// DroplessMLP_16535624089676
// MI455X (gfx1250) — matrix operand order repaired and checked by reading, not compiled, not run on hardware
//
#include <hip/hip_runtime.h>

// MoE GLU MLP for MI455X (gfx1250): token-gathered per-expert bf16 WMMA GEMMs.
// ws layout: [0..8) counts | [8..16) cursors | [16..25) padded CSR offsets |
//            [25] padded total | [32..32+MAXROWS) token ids | gates |
//            x1 bf16 buffer at byte offset 256KB (MAXROWS x F ~ 71MB).

#define T_DIM 4096
#define H_DIM 1024
#define F_DIM 4096
#define E_NUM 8
#define K_TOP 2
#define BM 64
#define BN 128                                  // block N tile (both GEMMs)
#define BK 32
#define MAXROWS (T_DIM * K_TOP + E_NUM * BM)    // 8704 (counts padded per expert)
#define MAXMT (MAXROWS / BM)                    // 136 M-tiles (worst case)
#define LDSK 40                                 // padded K stride (halves) = 80B
#define X1_OFF (256 * 1024)

typedef __attribute__((ext_vector_type(16))) __bf16 v16bf;
typedef __attribute__((ext_vector_type(8)))  __bf16 v8bf;
typedef __attribute__((ext_vector_type(8)))  float  v8f;

union Frag { v16bf v; v8bf h[2]; };

__device__ __forceinline__ unsigned short f2bf(float f) {
  union { __bf16 b; unsigned short u; } cv;
  cv.b = (__bf16)f;                             // hardware v_cvt f32->bf16 (RNE)
  return cv.u;
}

__device__ __forceinline__ v8bf cvt8(float4 f0, float4 f1) {
  v8bf r;
  r[0] = (__bf16)f0.x; r[1] = (__bf16)f0.y; r[2] = (__bf16)f0.z; r[3] = (__bf16)f0.w;
  r[4] = (__bf16)f1.x; r[5] = (__bf16)f1.y; r[6] = (__bf16)f1.z; r[7] = (__bf16)f1.w;
  return r;
}

__device__ __forceinline__ v8f wmma_bf16(v16bf a, v16bf b, v8f c) {
  return __builtin_amdgcn_wmma_f32_16x16x32_bf16(
      /*neg_a=*/false, a, /*neg_b=*/false, b,
      /*c_mod=*/(short)0, c, /*reuse_a=*/false, /*reuse_b=*/false);
}

// ---------------- setup kernels ----------------

__global__ void moe_init(float* out, int* wsI, int* tok, float* garr) {
  long long i = (long long)blockIdx.x * blockDim.x + threadIdx.x;
  if (i < (long long)T_DIM * H_DIM) out[i] = 0.0f;
  if (i < MAXROWS) { tok[i] = 0; garr[i] = 0.0f; }
  if (i < E_NUM) wsI[i] = 0;                    // counts
}

__global__ void moe_count(const int* __restrict__ top, int* wsI) {
  int i = blockIdx.x * blockDim.x + threadIdx.x;
  if (i < T_DIM * K_TOP) atomicAdd(&wsI[top[i]], 1);
}

__global__ void moe_scan(int* wsI) {
  if (blockIdx.x == 0 && threadIdx.x == 0) {
    int s = 0;
    for (int e = 0; e < E_NUM; e++) {
      wsI[16 + e] = s;
      s += (wsI[e] + BM - 1) & ~(BM - 1);       // pad each segment to BM
      wsI[8 + e] = 0;                           // scatter cursors
    }
    wsI[16 + E_NUM] = s;
    wsI[25] = s;                                // padded total rows
  }
}

__global__ void moe_scatter(const int* __restrict__ top,
                            const float* __restrict__ ew,
                            int* wsI, int* tok, float* garr) {
  int i = blockIdx.x * blockDim.x + threadIdx.x;
  if (i >= T_DIM * K_TOP) return;
  int e = top[i];
  int p = wsI[16 + e] + atomicAdd(&wsI[8 + e], 1);
  tok[p]  = i / K_TOP;
  garr[p] = ew[i];
}

// ---------------- GEMM1: x1 = silu(x*w1^T) .* (x*v1^T), bf16 out ----------------

__global__ __launch_bounds__(256) void moe_gemm1(
    const float* __restrict__ x, const float* __restrict__ w1,
    const float* __restrict__ v1, const int* __restrict__ wsI,
    const int* __restrict__ tok, unsigned short* __restrict__ x1) {
  __shared__ __align__(16) unsigned short sA [BM * LDSK];
  __shared__ __align__(16) unsigned short sB1[BN * LDSK];
  __shared__ __align__(16) unsigned short sB2[BN * LDSK];
  __shared__ int sTok[BM];

  const int tid   = threadIdx.x;
  const int rbase = blockIdx.y * BM;
  if (rbase >= wsI[25]) return;                 // beyond padded total
  int e = 0;
#pragma unroll
  for (int q = 1; q < E_NUM; q++) if (rbase >= wsI[16 + q]) e = q;
  const int fbase = blockIdx.x * BN;

  if (tid < BM) sTok[tid] = tok[rbase + tid];
  __syncthreads();

  const int lane  = tid & 31;
  const int wave  = tid >> 5;
  const int lhalf = lane >> 4;
  const int lmod  = lane & 15;
  const int m0    = (wave & 1) * 32;            // 2 waves in M
  const int n0    = (wave >> 1) * 32;           // 4 waves in N

  v8f acc1[2][2] = {};                          // [mi][nj]
  v8f acc2[2][2] = {};

  const int trA = tid >> 2;                     // A: 64 rows x 32 f32, 4 thr/row
  const int tcA = (tid & 3) * 8;
  const int trB = tid >> 1;                     // B: 128 rows x 32 f32, 2 thr/row
  const int tcB = (tid & 1) * 16;
  const float* pA  = x  + (size_t)sTok[trA] * H_DIM + tcA;
  const size_t wrow = ((size_t)e * F_DIM + fbase + trB) * H_DIM + tcB;
  const float* pB1 = w1 + wrow;
  const float* pB2 = v1 + wrow;

  for (int kk = 0; kk < H_DIM; kk += BK) {
    __builtin_prefetch(pB1 + kk + BK, 0, 1);
    __builtin_prefetch(pB2 + kk + BK, 0, 1);
    {  // A tile: gathered x rows, fp32 -> bf16
      float4 f0 = *(const float4*)(pA + kk);
      float4 f1 = *(const float4*)(pA + kk + 4);
      *(v8bf*)(sA + trA * LDSK + tcA) = cvt8(f0, f1);
    }
    {  // B1 tile: w1[e] rows are already Bt[n][k]
      float4 f0 = *(const float4*)(pB1 + kk);
      float4 f1 = *(const float4*)(pB1 + kk + 4);
      float4 f2 = *(const float4*)(pB1 + kk + 8);
      float4 f3 = *(const float4*)(pB1 + kk + 12);
      *(v8bf*)(sB1 + trB * LDSK + tcB)     = cvt8(f0, f1);
      *(v8bf*)(sB1 + trB * LDSK + tcB + 8) = cvt8(f2, f3);
    }
    {  // B2 tile: v1[e]
      float4 f0 = *(const float4*)(pB2 + kk);
      float4 f1 = *(const float4*)(pB2 + kk + 4);
      float4 f2 = *(const float4*)(pB2 + kk + 8);
      float4 f3 = *(const float4*)(pB2 + kk + 12);
      *(v8bf*)(sB2 + trB * LDSK + tcB)     = cvt8(f0, f1);
      *(v8bf*)(sB2 + trB * LDSK + tcB + 8) = cvt8(f2, f3);
    }
    __syncthreads();

    Frag a[2];  // A layout: lanes<16 K0-15, lanes>=16 K16-31 (matches B)
#pragma unroll
    for (int mi = 0; mi < 2; mi++) {
      const int mr = m0 + mi * 16 + lmod;
      a[mi].h[0] = *(const v8bf*)(sA + mr * LDSK + lhalf * 16);
      a[mi].h[1] = *(const v8bf*)(sA + mr * LDSK + lhalf * 16 + 8);
    }
#pragma unroll
    for (int nj = 0; nj < 2; nj++) {
      const int nn = n0 + nj * 16 + lmod;
      Frag b1, b2;  // B layout (same k order as A): lanes<16 K0-15, lanes>=16 K16-31
      b1.h[0] = *(const v8bf*)(sB1 + nn * LDSK + lhalf * 16);
      b1.h[1] = *(const v8bf*)(sB1 + nn * LDSK + lhalf * 16 + 8);
      b2.h[0] = *(const v8bf*)(sB2 + nn * LDSK + lhalf * 16);
      b2.h[1] = *(const v8bf*)(sB2 + nn * LDSK + lhalf * 16 + 8);
#pragma unroll
      for (int mi = 0; mi < 2; mi++) {
        acc1[mi][nj] = wmma_bf16(a[mi].v, b1.v, acc1[mi][nj]);
        acc2[mi][nj] = wmma_bf16(a[mi].v, b2.v, acc2[mi][nj]);
      }
    }
    __syncthreads();
  }

  // GLU epilogue: x1 = silu(c1) * c2  (C layout: vgpr i -> row i / i+8)
#pragma unroll
  for (int nj = 0; nj < 2; nj++) {
    const int col = fbase + n0 + nj * 16 + lmod;
#pragma unroll
    for (int mi = 0; mi < 2; mi++) {
#pragma unroll
      for (int i = 0; i < 8; i++) {
        const int rl = m0 + mi * 16 + i + lhalf * 8;
        float c1 = acc1[mi][nj][i], c2 = acc2[mi][nj][i];
        float s  = c1 * __builtin_amdgcn_rcpf(1.0f + __expf(-c1));  // fast silu
        x1[(size_t)(rbase + rl) * F_DIM + col] = f2bf(s * c2);
      }
    }
  }
}

// ---------------- GEMM2: out[tok] += gate * (x1 * w2) ----------------

__global__ __launch_bounds__(256) void moe_gemm2(
    const unsigned short* __restrict__ x1, const float* __restrict__ w2,
    const int* __restrict__ wsI, const int* __restrict__ tok,
    const float* __restrict__ garr, float* __restrict__ out) {
  __shared__ __align__(16) unsigned short sA[BM * LDSK];
  __shared__ __align__(16) unsigned short sB[BN * LDSK];
  __shared__ int   sTok[BM];
  __shared__ float sGate[BM];

  const int tid   = threadIdx.x;
  const int rbase = blockIdx.y * BM;
  if (rbase >= wsI[25]) return;
  int e = 0;
#pragma unroll
  for (int q = 1; q < E_NUM; q++) if (rbase >= wsI[16 + q]) e = q;
  const int hbase = blockIdx.x * BN;

  if (tid < BM) { sTok[tid] = tok[rbase + tid]; sGate[tid] = garr[rbase + tid]; }
  __syncthreads();

  const int lane  = tid & 31;
  const int wave  = tid >> 5;
  const int lhalf = lane >> 4;
  const int lmod  = lane & 15;
  const int m0    = (wave & 1) * 32;
  const int n0    = (wave >> 1) * 32;

  v8f acc[2][2] = {};

  const int trA = tid >> 2;                     // A: 64 rows x 32 halves
  const int tcA = (tid & 3) * 8;
  const unsigned short* pA = x1 + (size_t)(rbase + trA) * F_DIM + tcA;
  const int fl = tid >> 3;                      // B: 32 f-rows x 128 h-cols
  const int hc = (tid & 7) * 16;
  const float* pB = w2 + ((size_t)e * F_DIM + fl) * H_DIM + hbase + hc;

  for (int kk = 0; kk < F_DIM; kk += BK) {
    __builtin_prefetch(pB + (size_t)(kk + BK) * H_DIM, 0, 1);
    *(uint4*)(sA + trA * LDSK + tcA) = *(const uint4*)(pA + kk);  // bf16 direct
    {  // w2[f][h] -> transposed Bt[h][f] with fp32->bf16
      float4 f0 = *(const float4*)(pB + (size_t)kk * H_DIM);
      float4 f1 = *(const float4*)(pB + (size_t)kk * H_DIM + 4);
      float4 f2 = *(const float4*)(pB + (size_t)kk * H_DIM + 8);
      float4 f3 = *(const float4*)(pB + (size_t)kk * H_DIM + 12);
      unsigned short* d = sB + hc * LDSK + fl;
      d[ 0 * LDSK] = f2bf(f0.x); d[ 1 * LDSK] = f2bf(f0.y);
      d[ 2 * LDSK] = f2bf(f0.z); d[ 3 * LDSK] = f2bf(f0.w);
      d[ 4 * LDSK] = f2bf(f1.x); d[ 5 * LDSK] = f2bf(f1.y);
      d[ 6 * LDSK] = f2bf(f1.z); d[ 7 * LDSK] = f2bf(f1.w);
      d[ 8 * LDSK] = f2bf(f2.x); d[ 9 * LDSK] = f2bf(f2.y);
      d[10 * LDSK] = f2bf(f2.z); d[11 * LDSK] = f2bf(f2.w);
      d[12 * LDSK] = f2bf(f3.x); d[13 * LDSK] = f2bf(f3.y);
      d[14 * LDSK] = f2bf(f3.z); d[15 * LDSK] = f2bf(f3.w);
    }
    __syncthreads();

    Frag a[2];
#pragma unroll
    for (int mi = 0; mi < 2; mi++) {
      const int mr = m0 + mi * 16 + lmod;
      a[mi].h[0] = *(const v8bf*)(sA + mr * LDSK + lhalf * 16);
      a[mi].h[1] = *(const v8bf*)(sA + mr * LDSK + lhalf * 16 + 8);
    }
#pragma unroll
    for (int nj = 0; nj < 2; nj++) {
      const int nn = n0 + nj * 16 + lmod;
      Frag b;
      b.h[0] = *(const v8bf*)(sB + nn * LDSK + lhalf * 16);
      b.h[1] = *(const v8bf*)(sB + nn * LDSK + lhalf * 16 + 8);
#pragma unroll
      for (int mi = 0; mi < 2; mi++)
        acc[mi][nj] = wmma_bf16(a[mi].v, b.v, acc[mi][nj]);
    }
    __syncthreads();
  }

  // gate-scaled scatter-add (padding rows: gate==0 -> adds 0)
#pragma unroll
  for (int nj = 0; nj < 2; nj++) {
    const int col = hbase + n0 + nj * 16 + lmod;
#pragma unroll
    for (int mi = 0; mi < 2; mi++) {
#pragma unroll
      for (int i = 0; i < 8; i++) {
        const int rl = m0 + mi * 16 + i + lhalf * 8;
        atomicAdd(&out[(size_t)sTok[rl] * H_DIM + col], acc[mi][nj][i] * sGate[rl]);
      }
    }
  }
}

// ---------------- launcher ----------------

extern "C" void kernel_launch(void* const* d_in, const int* in_sizes, int n_in,
                              void* d_out, int out_size, void* d_ws, size_t ws_size,
                              hipStream_t stream) {
  const float* x  = (const float*)d_in[0];
  // d_in[1] = scores (unused by reference math)
  const float* ew = (const float*)d_in[2];
  const int*   te = (const int*)d_in[3];
  const float* w1 = (const float*)d_in[4];
  const float* v1 = (const float*)d_in[5];
  const float* w2 = (const float*)d_in[6];
  float* out = (float*)d_out;

  int*   wsI  = (int*)d_ws;
  int*   tok  = wsI + 32;
  float* garr = (float*)(wsI + 32 + MAXROWS);
  unsigned short* x1 = (unsigned short*)((char*)d_ws + X1_OFF);

  const int nInit = T_DIM * H_DIM;
  moe_init   <<<(nInit + 255) / 256, 256, 0, stream>>>(out, wsI, tok, garr);
  moe_count  <<<(T_DIM * K_TOP + 255) / 256, 256, 0, stream>>>(te, wsI);
  moe_scan   <<<1, 32, 0, stream>>>(wsI);
  moe_scatter<<<(T_DIM * K_TOP + 255) / 256, 256, 0, stream>>>(te, ew, wsI, tok, garr);
  moe_gemm1  <<<dim3(F_DIM / BN, MAXMT), 256, 0, stream>>>(x, w1, v1, wsI, tok, x1);
  moe_gemm2  <<<dim3(H_DIM / BN, MAXMT), 256, 0, stream>>>(x1, w2, wsI, tok, garr, out);
}
